// RNNAttentionDecoder_12515534700793
// MI455X (gfx1250) — matrix operand order repaired and checked by reading, not compiled, not run on hardware
//
#include <hip/hip_runtime.h>
#include <hip/hip_bf16.h>

// ---------------------------------------------------------------------------
// RNN attention decoder for MI455X (gfx1250, wave32, WMMA)
// GEMMs on v_wmma_f32_16x16x32_bf16, fp32 accumulate, fragment-order LDS.
// ---------------------------------------------------------------------------

typedef __attribute__((ext_vector_type(16))) __bf16 v16bf;
typedef __attribute__((ext_vector_type(8)))  float  v8f;

#define T_STEPS 32
#define BATCH   128
#define SRC     64
#define HID     1024
#define EWD     512
#define EFD     64
#define VWD     32000
#define VFD     512

__device__ __forceinline__ unsigned int f2bf(float f) {
  unsigned int u = __builtin_bit_cast(unsigned int, f);
  unsigned int r = u + 0x7FFFu + ((u >> 16) & 1u);   // round-to-nearest-even
  return r >> 16;
}
__device__ __forceinline__ unsigned int pack2bf(float lo, float hi) {
  return f2bf(lo) | (f2bf(hi) << 16);
}

// ---------------------------------------------------------------------------
// C(M,N) = act( A(M,K) @ W(N,K)^T + bias )    tile 64x64 per block (M=128).
// Block = 128 threads = 4 waves; each wave owns a 16x64 strip (4 wmma accs).
// LDS holds tiles pre-swizzled into WMMA fragment order:
//   lAf: wave w, lane l -> 16 contiguous bf16 = that lane's A fragment
//   lBf: tile j, lane l -> 16 contiguous bf16 = that lane's B fragment
// so each fragment load is 2x ds_load_b128 (no scalar gathers).
// ---------------------------------------------------------------------------
__global__ __launch_bounds__(128)
void gemm_bt_wmma(const float* __restrict__ A, int lda,
                  const float* __restrict__ W, int ldw,
                  const float* __restrict__ bias,
                  float* __restrict__ C, int ldc,
                  int K, int act)
{
  __shared__ unsigned int lAf[4 * 32 * 8];  // 4 waves * 32 lanes * 16 bf16
  __shared__ unsigned int lBf[4 * 32 * 8];  // 4 j-tiles * 32 lanes * 16 bf16

  const int tid  = threadIdx.x;
  const int lane = tid & 31;
  const int wave = tid >> 5;
  const int half = lane >> 4;
  const int l16  = lane & 15;

  const int n0 = blockIdx.x * 64;
  const int m0 = blockIdx.y * 64;

  // staging role: each thread stages 16 consecutive k of one row, per tile
  const int srow = tid >> 1;          // 0..63
  const int skb  = (tid & 1) * 16;    // 0 or 16

  v8f acc[4] = {};

  for (int k0 = 0; k0 < K; k0 += 32) {
    // ---- stage A tile row (fp32 -> bf16, swizzled to fragment order) ----
    {
      const float4* s4 = reinterpret_cast<const float4*>(
          A + (long)(m0 + srow) * lda + k0 + skb);
      float4 v0 = s4[0], v1 = s4[1], v2 = s4[2], v3 = s4[3];
      float vv[16] = {v0.x,v0.y,v0.z,v0.w, v1.x,v1.y,v1.z,v1.w,
                      v2.x,v2.y,v2.z,v2.w, v3.x,v3.y,v3.z,v3.w};
      const int w = srow >> 4, al16 = srow & 15;
#pragma unroll
      for (int g = 0; g < 2; ++g) {            // two 8-wide k sub-groups
        const int kloc  = skb + g * 8;         // 0,8 or 16,24
        const int h     = (kloc >> 4) & 1;     // dest lane half
        const int ibase = ((kloc >> 3) & 1) * 8; // frag element base (0 or 8)
        unsigned int* dst = &lAf[(w * 32 + al16 + 16 * h) * 8 + (ibase >> 1)];
#pragma unroll
        for (int p = 0; p < 4; ++p)
          dst[p] = pack2bf(vv[g * 8 + 2 * p], vv[g * 8 + 2 * p + 1]);
      }
    }
    // ---- stage B tile row (W row n -> fragment order) ----
    {
      const float4* s4 = reinterpret_cast<const float4*>(
          W + (long)(n0 + srow) * ldw + k0 + skb);
      float4 v0 = s4[0], v1 = s4[1], v2 = s4[2], v3 = s4[3];
      float vv[16] = {v0.x,v0.y,v0.z,v0.w, v1.x,v1.y,v1.z,v1.w,
                      v2.x,v2.y,v2.z,v2.w, v3.x,v3.y,v3.z,v3.w};
      const int j = srow >> 4, bl16 = srow & 15;
      const int h = skb >> 4;                  // dest lane half from k range
      unsigned int* dst = &lBf[(j * 32 + bl16 + 16 * h) * 8];
#pragma unroll
      for (int p = 0; p < 8; ++p)
        dst[p] = pack2bf(vv[2 * p], vv[2 * p + 1]);
    }
    __syncthreads();

    // ---- fragments are contiguous per lane: 2x ds_load_b128 each ----
    v16bf af = *reinterpret_cast<const v16bf*>(&lAf[(wave * 32 + lane) * 8]);
#pragma unroll
    for (int j = 0; j < 4; ++j) {
      v16bf bfrag = *reinterpret_cast<const v16bf*>(&lBf[(j * 32 + lane) * 8]);
      acc[j] = __builtin_amdgcn_wmma_f32_16x16x32_bf16(
          false, af, false, bfrag, (short)0, acc[j], false, false);
    }
    __syncthreads();
  }

  // ---- epilogue: bias + optional tanh (C layout: VGPR r -> M = r + 8*half) ----
#pragma unroll
  for (int j = 0; j < 4; ++j) {
    int n = n0 + j * 16 + l16;
    float bv = bias ? bias[n] : 0.0f;
#pragma unroll
    for (int r = 0; r < 8; ++r) {
      int m = m0 + wave * 16 + r + 8 * half;
      float v = acc[j][r] + bv;
      if (act) v = tanhf(v);
      C[(long)m * ldc + n] = v;
    }
  }
}

// ---------------------------------------------------------------------------
// x = concat(word_emb[wid], field_emb[fid], prev_out)   (B, 1600)
// ---------------------------------------------------------------------------
__global__ void build_x(const int* __restrict__ wid, const int* __restrict__ fid,
                        const float* __restrict__ wtab, const float* __restrict__ ftab,
                        const float* __restrict__ prev, float* __restrict__ x)
{
  int idx = blockIdx.x * blockDim.x + threadIdx.x;
  if (idx >= BATCH * (EWD + EFD + HID)) return;
  int b = idx / (EWD + EFD + HID);
  int c = idx % (EWD + EFD + HID);
  float v;
  if (c < EWD)            v = wtab[(long)wid[b] * EWD + c];
  else if (c < EWD + EFD) v = ftab[(long)fid[b] * EFD + (c - EWD)];
  else                    v = prev[b * HID + (c - EWD - EFD)];
  x[idx] = v;
}

// ---------------------------------------------------------------------------
// GRU gates (in-place hidden update): h = (1-z)*n + z*h
// ---------------------------------------------------------------------------
__global__ void gru_gate(const float* __restrict__ gi, const float* __restrict__ gh,
                         float* __restrict__ h)
{
  int idx = blockIdx.x * blockDim.x + threadIdx.x;
  if (idx >= BATCH * HID) return;
  int b = idx / HID, c = idx % HID;
  long base = (long)b * 3 * HID;
  float ir = gi[base + c],            hr = gh[base + c];
  float iz = gi[base + HID + c],      hz = gh[base + HID + c];
  float in_ = gi[base + 2 * HID + c], hn = gh[base + 2 * HID + c];
  float r = 1.0f / (1.0f + expf(-(ir + hr)));
  float z = 1.0f / (1.0f + expf(-(iz + hz)));
  float n = tanhf(in_ + r * hn);
  h[idx] = (1.0f - z) * n + z * h[idx];
}

// ---------------------------------------------------------------------------
// Global attention: scores = q . context, masked softmax, ctx = attn . context
// ---------------------------------------------------------------------------
__global__ __launch_bounds__(256)
void attn_ctx(const float* __restrict__ q, const float* __restrict__ context,
              const unsigned char* __restrict__ mask, float* __restrict__ ctx)
{
  int b = blockIdx.x;
  int tid = threadIdx.x;
  __shared__ float part[SRC][4];
  __shared__ float sc[SRC];
  __shared__ float aw[SRC];

  const float* qb = q + (long)b * HID;
  const float* cb = context + (long)b * SRC * HID;

  int s = tid >> 2, quarter = tid & 3;
  float acc = 0.0f;
  for (int h = quarter * 256; h < (quarter + 1) * 256; ++h)
    acc += qb[h] * cb[(long)s * HID + h];
  part[s][quarter] = acc;
  __syncthreads();

  if (tid < SRC) {
    float v = part[tid][0] + part[tid][1] + part[tid][2] + part[tid][3];
    sc[tid] = mask[b * SRC + tid] ? v : -1e9f;
  }
  __syncthreads();

  float mx = -1e30f;
  for (int i = 0; i < SRC; ++i) mx = fmaxf(mx, sc[i]);
  if (tid < SRC) aw[tid] = expf(sc[tid] - mx);
  __syncthreads();

  float sum = 0.0f;
  for (int i = 0; i < SRC; ++i) sum += aw[i];
  float inv = 1.0f / sum;

  for (int h = tid; h < HID; h += 256) {
    float a = 0.0f;
    for (int s2 = 0; s2 < SRC; ++s2)
      a += aw[s2] * cb[(long)s2 * HID + h];
    ctx[(long)b * HID + h] = a * inv;
  }
}

// ---------------------------------------------------------------------------
__global__ void cat2(const float* __restrict__ c0, const float* __restrict__ c1,
                     float* __restrict__ cat)
{
  int idx = blockIdx.x * blockDim.x + threadIdx.x;
  if (idx >= BATCH * 2 * HID) return;
  int b = idx / (2 * HID), c = idx % (2 * HID);
  cat[idx] = (c < HID) ? c0[b * HID + c] : c1[b * HID + (c - HID)];
}

__global__ void copy_f(const float* __restrict__ src, float* __restrict__ dst, int n)
{
  int i = blockIdx.x * blockDim.x + threadIdx.x;
  if (i < n) dst[i] = src[i];
}

// ---------------------------------------------------------------------------
extern "C" void kernel_launch(void* const* d_in, const int* in_sizes, int n_in,
                              void* d_out_v, int out_size, void* d_ws, size_t ws_size,
                              hipStream_t stream)
{
  (void)in_sizes; (void)n_in; (void)out_size; (void)ws_size;

  const int*   word_ids  = (const int*)d_in[0];
  const int*   field_ids = (const int*)d_in[1];
  /* d_in[2] = lengths (unused in forward math) */
  const float* hidden    = (const float*)d_in[3];
  const float* context   = (const float*)d_in[4];
  const unsigned char* cmask = (const unsigned char*)d_in[5];
  const float* prev_out  = (const float*)d_in[6];
  const float* wtab      = (const float*)d_in[7];
  const float* ftab      = (const float*)d_in[8];
  const float* g0Wih = (const float*)d_in[9];
  const float* g0Whh = (const float*)d_in[10];
  const float* g0bih = (const float*)d_in[11];
  const float* g0bhh = (const float*)d_in[12];
  const float* g1Wih = (const float*)d_in[13];
  const float* g1Whh = (const float*)d_in[14];
  const float* g1bih = (const float*)d_in[15];
  const float* g1bhh = (const float*)d_in[16];
  const float* Wa = (const float*)d_in[17];
  const float* Wc = (const float*)d_in[18];
  const float* Ww = (const float*)d_in[19];
  const float* bw = (const float*)d_in[20];
  const float* Wf = (const float*)d_in[21];
  const float* bfv = (const float*)d_in[22];

  float* out = (float*)d_out_v;
  float* ws  = (float*)d_ws;

  // workspace layout (floats)
  float* h0  = ws;                        // B*H
  float* h1  = h0 + BATCH * HID;          // B*H   (h0,h1 contiguous = stacked hidden)
  float* ob  = h1 + BATCH * HID;          // B*H   (input-feeding output)
  float* x   = ob + BATCH * HID;          // B*1600
  float* gi  = x  + BATCH * (EWD + EFD + HID);  // B*3072
  float* gh  = gi + BATCH * 3 * HID;      // B*3072
  float* qb  = gh + BATCH * 3 * HID;      // B*H
  float* cx  = qb + BATCH * HID;          // B*H
  float* cat = cx + BATCH * HID;          // B*2H

  const long O1 = (long)T_STEPS * BATCH * VWD;       // field_scores offset
  const long O2 = O1 + (long)T_STEPS * BATCH * VFD;  // hidden offset
  const long O3 = O2 + 2L * BATCH * HID;             // output offset

  // init carried state
  copy_f<<<(2 * BATCH * HID + 255) / 256, 256, 0, stream>>>(hidden, h0, 2 * BATCH * HID);
  copy_f<<<(BATCH * HID + 255) / 256, 256, 0, stream>>>(prev_out, ob, BATCH * HID);

  for (int t = 0; t < T_STEPS; ++t) {
    build_x<<<(BATCH * (EWD + EFD + HID) + 255) / 256, 256, 0, stream>>>(
        word_ids + t * BATCH, field_ids + t * BATCH, wtab, ftab, ob, x);

    // GRU layer 0
    gemm_bt_wmma<<<dim3(3 * HID / 64, BATCH / 64), 128, 0, stream>>>(
        x, EWD + EFD + HID, g0Wih, EWD + EFD + HID, g0bih, gi, 3 * HID, EWD + EFD + HID, 0);
    gemm_bt_wmma<<<dim3(3 * HID / 64, BATCH / 64), 128, 0, stream>>>(
        h0, HID, g0Whh, HID, g0bhh, gh, 3 * HID, HID, 0);
    gru_gate<<<(BATCH * HID + 255) / 256, 256, 0, stream>>>(gi, gh, h0);

    // GRU layer 1
    gemm_bt_wmma<<<dim3(3 * HID / 64, BATCH / 64), 128, 0, stream>>>(
        h0, HID, g1Wih, HID, g1bih, gi, 3 * HID, HID, 0);
    gemm_bt_wmma<<<dim3(3 * HID / 64, BATCH / 64), 128, 0, stream>>>(
        h1, HID, g1Whh, HID, g1bhh, gh, 3 * HID, HID, 0);
    gru_gate<<<(BATCH * HID + 255) / 256, 256, 0, stream>>>(gi, gh, h1);

    // attention
    gemm_bt_wmma<<<dim3(HID / 64, BATCH / 64), 128, 0, stream>>>(
        h1, HID, Wa, HID, nullptr, qb, HID, HID, 0);
    attn_ctx<<<BATCH, 256, 0, stream>>>(qb, context, cmask, cx);

    // out = tanh([ctx, h1] @ Wc^T)
    cat2<<<(BATCH * 2 * HID + 255) / 256, 256, 0, stream>>>(cx, h1, cat);
    gemm_bt_wmma<<<dim3(HID / 64, BATCH / 64), 128, 0, stream>>>(
        cat, 2 * HID, Wc, 2 * HID, nullptr, ob, HID, 2 * HID, 1 /*tanh*/);

    // generator -> d_out directly
    gemm_bt_wmma<<<dim3(VWD / 64, BATCH / 64), 128, 0, stream>>>(
        ob, HID, Ww, HID, bw, out + (long)t * BATCH * VWD, VWD, HID, 0);
    gemm_bt_wmma<<<dim3(VFD / 64, BATCH / 64), 128, 0, stream>>>(
        ob, HID, Wf, HID, bfv, out + O1 + (long)t * BATCH * VFD, VFD, HID, 0);
  }

  // final hidden (stacked h0,h1) and output
  copy_f<<<(2 * BATCH * HID + 255) / 256, 256, 0, stream>>>(h0, out + O2, 2 * BATCH * HID);
  copy_f<<<(BATCH * HID + 255) / 256, 256, 0, stream>>>(ob, out + O3, BATCH * HID);
}
